// PrototypeConditionedUnitDeltaWriterV3440_84791244358062
// MI455X (gfx1250) — matrix operand order repaired and checked by reading, not compiled, not run on hardware
//
#include <hip/hip_runtime.h>
#include <hip/hip_bf16.h>

// ---------------------------------------------------------------------------
// Problem dimensions (compile-time constants from the reference)
// ---------------------------------------------------------------------------
#define B_    4
#define M_    512
#define U_    128
#define H_    8
#define K_    4
#define D_    768
#define T_    512
#define C_    256
#define P_    256
#define QIN_  1027          // 4C+3
#define QINP_ 1056          // padded to multiple of 32
#define FIN_  1284          // 5C+4
#define FINP_ 1312          // padded to multiple of 32
#define ROWS_ 4096          // B*U*H
#define MK_   2048          // M*K
#define BH_   32            // B*H
#define KVROWS_ 65536       // B*M*H*K

typedef __bf16 bf16;
typedef __attribute__((ext_vector_type(8)))  __bf16 bf16x8;
typedef __attribute__((ext_vector_type(16))) __bf16 bf16x16;
typedef __attribute__((ext_vector_type(8)))  float  f32x8;

// ---------------------------------------------------------------------------
// Device helpers
// ---------------------------------------------------------------------------
__device__ __forceinline__ float nan2num(float v) {
    if (v != v) return 0.0f;
    if (v >  3.4028235e38f) return  3.4028235e38f;
    if (v < -3.4028235e38f) return -3.4028235e38f;
    return v;
}

__device__ __forceinline__ float gelu_exact(float x) {
    return 0.5f * x * (1.0f + erff(x * 0.70710678118654752f));
}

__device__ __forceinline__ float block_sum(float v, float* red) {
    int t = threadIdx.x;
    red[t] = v; __syncthreads();
    for (int off = 128; off > 0; off >>= 1) {
        if (t < off) red[t] += red[t + off];
        __syncthreads();
    }
    float r = red[0]; __syncthreads();
    return r;
}

__device__ __forceinline__ float block_max(float v, float* red) {
    int t = threadIdx.x;
    red[t] = v; __syncthreads();
    for (int off = 128; off > 0; off >>= 1) {
        if (t < off) red[t] = fmaxf(red[t], red[t + off]);
        __syncthreads();
    }
    float r = red[0]; __syncthreads();
    return r;
}

// ---------------------------------------------------------------------------
// Workhorse: batched bf16 WMMA GEMM, 64x64 macro-tile, K-step 32.
//   C = epilogue( A[Mr,Kr] @ Bt^T ),  Bt stored [Nr,Kr] row-major ("B^T").
// All B-operands in this pipeline are pre-transposed so LDS staging is pure
// b128 traffic (no scalar-transpose scatter).
//   epilogue: optional row-scale, bias, exact GELU; f32 and/or bf16 writes,
//   bf16 write optionally transposed (for the value matrix).
// Requires: Mr%64==0, Nr%64==0, Kr%32==0, 16B-aligned buffers.
// 8 waves per block: wave (wm,wn) owns a 16x32 piece (two 16x16 accumulators).
// ---------------------------------------------------------------------------
#define LDSP 40   // LDS row pitch (elements); 80B rows keep 16B alignment

__global__ __launch_bounds__(256) void wmma_gemm(
    const bf16*  __restrict__ A,  long strideA,
    const bf16*  __restrict__ Bt, long strideB,
    const float* __restrict__ bias,
    const float* __restrict__ rscale, long strideR,
    float* __restrict__ Cf, long strideCf,
    bf16*  __restrict__ Cb, long strideCb,
    int Mr, int Nr, int Kr, int transCb, int gelu_flag)
{
    __shared__ __align__(16) bf16 sA[64 * LDSP];
    __shared__ __align__(16) bf16 sB[64 * LDSP];   // [n][k]

    const int z   = blockIdx.z;
    const bf16* Ab = A  + (long)z * strideA;
    const bf16* Bb = Bt + (long)z * strideB;
    const int m0 = blockIdx.y * 64;
    const int n0 = blockIdx.x * 64;
    const int tid  = threadIdx.x;
    const int lane = tid & 31;
    const int wave = tid >> 5;
    const int wm = wave & 3;        // 0..3 -> 16-row strip
    const int wn = wave >> 2;       // 0..1 -> 32-col strip

    // staging coordinates (same shape for A and Bt: 64 rows x 32 K)
    const int sr = tid >> 2;        // 0..63
    const int sc = (tid & 3) * 8;   // 0,8,16,24

    f32x8 acc0 = {0.f,0.f,0.f,0.f,0.f,0.f,0.f,0.f};
    f32x8 acc1 = {0.f,0.f,0.f,0.f,0.f,0.f,0.f,0.f};

    for (int k0 = 0; k0 < Kr; k0 += 32) {
        // ---- stage A tile 64x32 and B^T tile 64x32 (both b128) ----
        const bf16* srcA = Ab + (long)(m0 + sr) * Kr + k0 + sc;
        const bf16* srcB = Bb + (long)(n0 + sr) * Kr + k0 + sc;
        *(bf16x8*)&sA[sr * LDSP + sc] = *(const bf16x8*)srcA;
        *(bf16x8*)&sB[sr * LDSP + sc] = *(const bf16x8*)srcB;
        if (k0 + 32 < Kr) {                       // next-slice prefetch
            __builtin_prefetch(srcA + 32, 0, 1);  // -> global_prefetch_b8
            __builtin_prefetch(srcB + 32, 0, 1);
        }
        __syncthreads();

        // ---- A fragment: row m, K-contiguous chunk of 16 (k = 16h + s) ----
        {
            int am = lane & 15, ah = lane >> 4;
            const bf16* arow = &sA[(wm * 16 + am) * LDSP];
            bf16x8 a_lo = *(const bf16x8*)(arow + ah * 16);
            bf16x8 a_hi = *(const bf16x8*)(arow + ah * 16 + 8);
            bf16x16 afrag = __builtin_shufflevector(a_lo, a_hi,
                0,1,2,3,4,5,6,7,8,9,10,11,12,13,14,15);

            // ---- B fragments: column n, K-contiguous chunk of 16 ----
            int bn = lane & 15, bh = lane >> 4;
            const bf16* b0p = &sB[(wn * 32 + bn) * LDSP + bh * 16];
            bf16x8 b0lo = *(const bf16x8*)(b0p);
            bf16x8 b0hi = *(const bf16x8*)(b0p + 8);
            bf16x16 bfrag0 = __builtin_shufflevector(b0lo, b0hi,
                0,1,2,3,4,5,6,7,8,9,10,11,12,13,14,15);
            const bf16* b1p = &sB[(wn * 32 + 16 + bn) * LDSP + bh * 16];
            bf16x8 b1lo = *(const bf16x8*)(b1p);
            bf16x8 b1hi = *(const bf16x8*)(b1p + 8);
            bf16x16 bfrag1 = __builtin_shufflevector(b1lo, b1hi,
                0,1,2,3,4,5,6,7,8,9,10,11,12,13,14,15);

            acc0 = __builtin_amdgcn_wmma_f32_16x16x32_bf16(
                false, afrag, false, bfrag0, (short)0, acc0, false, false);
            acc1 = __builtin_amdgcn_wmma_f32_16x16x32_bf16(
                false, afrag, false, bfrag1, (short)0, acc1, false, false);
        }
        __syncthreads();
    }

    // ---- epilogue: C layout (v8f): lanes0-15 M=r, lanes16-31 M=8+r; N=lane%16
    const int cn = lane & 15;
    const int ch = lane >> 4;
    #pragma unroll
    for (int half = 0; half < 2; ++half) {
        f32x8 acc = half ? acc1 : acc0;
        int col = n0 + wn * 32 + half * 16 + cn;
        #pragma unroll
        for (int r = 0; r < 8; ++r) {
            int row = m0 + wm * 16 + ch * 8 + r;
            float x = acc[r];
            if (rscale) x *= rscale[(long)z * strideR + row];
            if (bias)   x += bias[col];
            if (gelu_flag) x = gelu_exact(x);
            if (Cf) Cf[(long)z * strideCf + (long)row * Nr + col] = x;
            if (Cb) {
                long o = transCb ? ((long)col * Mr + row) : ((long)row * Nr + col);
                Cb[(long)z * strideCb + o] = (bf16)x;
            }
        }
    }
}

// ---------------------------------------------------------------------------
// Elementwise / prep kernels
// ---------------------------------------------------------------------------
// transpose-pack weights: src f32 [rows, cols] -> dst bf16 [cols, prows]
// (prows >= rows, zero padded) so GEMM consumes them as B^T with K-contiguity.
__global__ __launch_bounds__(256) void k_pack_wT(
    const float* __restrict__ src, bf16* __restrict__ dst,
    int rows, int cols, int prows)
{
    long n = (long)cols * prows;
    long i = (long)blockIdx.x * 256 + threadIdx.x;
    long stride = (long)gridDim.x * 256;
    for (; i < n; i += stride) {
        int c = (int)(i / prows);
        int r = (int)(i % prows);
        dst[i] = (r < rows) ? (bf16)src[(long)r * cols + c] : (bf16)0.0f;
    }
}

// future_trace [B,M,H*T] -> bf16 transposed [B, H*T, M]
__global__ __launch_bounds__(256) void k_ftT(
    const float* __restrict__ ft, bf16* __restrict__ dst)
{
    long n = (long)B_ * (H_ * T_) * M_;
    long i = (long)blockIdx.x * 256 + threadIdx.x;
    long stride = (long)gridDim.x * 256;
    for (; i < n; i += stride) {
        int m  = (int)(i % M_);
        int nn = (int)((i / M_) % (H_ * T_));
        int b  = (int)(i / ((long)M_ * H_ * T_));
        dst[i] = (bf16)ft[((long)b * M_ + m) * (H_ * T_) + nn];
    }
}

// assign [B,M,U] -> assignT bf16 [B,U,M]  (A operand of pooling GEMMs)
__global__ __launch_bounds__(256) void k_assignT(
    const float* __restrict__ a, bf16* __restrict__ at)
{
    long n = (long)B_ * U_ * M_;
    long i = (long)blockIdx.x * 256 + threadIdx.x;
    if (i >= n) return;
    int m = (int)(i % M_);
    int u = (int)((i / M_) % U_);
    int b = (int)(i / ((long)M_ * U_));
    at[i] = (bf16)a[((long)b * M_ + m) * U_ + u];
}

// point_evidence, transposed: peT bf16 [B, H*D, M]
__global__ __launch_bounds__(256) void k_point_evidenceT(
    const float* __restrict__ raw, const float* __restrict__ tw,
    bf16* __restrict__ peT)
{
    long n = (long)B_ * (H_ * D_) * M_;
    long i = (long)blockIdx.x * 256 + threadIdx.x;
    long stride = (long)gridDim.x * 256;
    for (; i < n; i += stride) {
        int m  = (int)(i % M_);
        int hd = (int)((i / M_) % (H_ * D_));
        int b  = (int)(i / ((long)M_ * H_ * D_));
        int h = hd / D_, d = hd % D_;
        long bm = (long)b * M_ + m;
        float s = 0.f;
        #pragma unroll
        for (int k = 0; k < K_; ++k) {
            long rk = (bm * H_ + h) * K_ + k;
            s += nan2num(raw[rk * D_ + d]) * fmaxf(tw[rk], 1e-8f);
        }
        peT[i] = (bf16)s;
    }
}

// anchor, transposed: anT bf16 [B, D, M] = mean over 4 of nan2num(obs_sem)
__global__ __launch_bounds__(256) void k_anchorT(
    const float* __restrict__ obs, bf16* __restrict__ anT)
{
    long n = (long)B_ * D_ * M_;
    long i = (long)blockIdx.x * 256 + threadIdx.x;
    long stride = (long)gridDim.x * 256;
    for (; i < n; i += stride) {
        int m = (int)(i % M_);
        int d = (int)((i / M_) % D_);
        int b = (int)(i / ((long)M_ * D_));
        long bm = (long)b * M_ + m;
        float s = 0.f;
        #pragma unroll
        for (int j = 0; j < 4; ++j)
            s += nan2num(obs[(bm * 4 + j) * (long)D_ + d]);
        anT[i] = (bf16)(s * 0.25f);
    }
}

// den_inv[B,U], sem_usage_c[B,U,H], mass_c[B,U,H]
__global__ __launch_bounds__(256) void k_pools(
    const float* __restrict__ assign, const float* __restrict__ psu,
    float* __restrict__ den_inv, float* __restrict__ semu,
    float* __restrict__ massc)
{
    __shared__ float red[256];
    int bu = blockIdx.x;
    int u = bu % U_, b = bu / U_;
    float s = 0.f;
    for (int m = threadIdx.x; m < M_; m += 256)
        s += assign[((long)b * M_ + m) * U_ + u];
    float den = fmaxf(block_sum(s, red), 1e-6f);
    if (threadIdx.x == 0) den_inv[bu] = 1.0f / den;
    for (int h = 0; h < H_; ++h) {
        float t = 0.f;
        for (int m = threadIdx.x; m < M_; m += 256)
            t += assign[((long)b * M_ + m) * U_ + u] *
                 psu[((long)b * M_ + m) * H_ + h];
        float tt = block_sum(t, red);
        if (threadIdx.x == 0) {
            semu[bu * H_ + h]  = fminf(fmaxf(tt / den, 0.f), 1.f);
            massc[bu * H_ + h] = fminf(fmaxf(fmaxf(tt, 0.f) / den, 0.f), 1.f);
        }
    }
}

// LayerNorm over rows -> bf16 (optionally zero-padded to ypitch).
// mode 0: y row = x row.  mode 1: x = topk_raw rows (b,m,h,k), nan_to_num,
//         y row permuted to (b,h, m*K+k) layout.
__global__ __launch_bounds__(256) void k_ln_rows(
    const float* __restrict__ x, const float* __restrict__ g,
    const float* __restrict__ bt, bf16* __restrict__ y,
    int Dn, int ypitch, int mode)
{
    __shared__ float red[256];
    int r = blockIdx.x;
    long xoff = (long)r * Dn;
    long yrow = r;
    bool n2n = false;
    if (mode == 1) {
        int k = r % K_;
        int h = (r / K_) % H_;
        int m = (r / (K_ * H_)) % M_;
        int b = r / (K_ * H_ * M_);
        yrow = (((long)(b * H_ + h) * M_ + m) * K_ + k);
        n2n = true;
    }
    float vbuf[8];
    int cnt = 0;
    float s = 0.f;
    for (int i = threadIdx.x; i < Dn; i += 256) {
        float v = x[xoff + i];
        if (n2n) v = nan2num(v);
        vbuf[cnt++] = v;
        s += v;
    }
    float mu = block_sum(s, red) / (float)Dn;
    float s2 = 0.f;
    cnt = 0;
    for (int i = threadIdx.x; i < Dn; i += 256) {
        float d = vbuf[cnt++] - mu;
        s2 += d * d;
    }
    float var = block_sum(s2, red) / (float)Dn;
    float rstd = rsqrtf(var + 1e-5f);
    cnt = 0;
    long yo = yrow * ypitch;
    for (int i = threadIdx.x; i < Dn; i += 256)
        y[yo + i] = (bf16)((vbuf[cnt++] - mu) * rstd * g[i] + bt[i]);
    for (int i = Dn + threadIdx.x; i < ypitch; i += 256)
        y[yo + i] = (bf16)0.0f;
}

// q_in f32 [ROWS_, QIN_]
__global__ __launch_bounds__(256) void k_concat_q(
    const float* __restrict__ trace_h, const float* __restrict__ anchor_h,
    const float* __restrict__ evid_h,  const float* __restrict__ old_h,
    const float* __restrict__ conf,    const float* __restrict__ usage,
    const float* __restrict__ semu,    float* __restrict__ qin)
{
    int row = blockIdx.x;
    int h = row % H_;
    int u = (row / H_) % U_;
    int b = row / (H_ * U_);
    int bu = b * U_ + u;
    long o = (long)row * QIN_;
    for (int i = threadIdx.x; i < QIN_; i += 256) {
        float v;
        if (i < 256)       v = trace_h[(long)row * C_ + i];
        else if (i < 512)  v = anchor_h[(long)bu * C_ + (i - 256)];
        else if (i < 768)  v = evid_h[(long)row * C_ + (i - 512)];
        else if (i < 1024) v = old_h[(long)row * C_ + (i - 768)];
        else if (i == 1024) v = conf[bu];
        else if (i == 1025) v = fminf(fmaxf(usage[bu * H_ + h], 0.f), 1.f);
        else                v = semu[bu * H_ + h];
        qin[o + i] = v;
    }
}

// query f32 [B,U,H,C] -> bf16 [B,H,U,C]
__global__ __launch_bounds__(256) void k_permute_query(
    const float* __restrict__ q, bf16* __restrict__ qp)
{
    long n = (long)ROWS_ * C_;
    long i = (long)blockIdx.x * 256 + threadIdx.x;
    if (i >= n) return;
    int c = (int)(i % C_);
    int row = (int)(i / C_);
    int h = row % H_;
    int u = (row / H_) % U_;
    int b = row / (H_ * U_);
    qp[(((long)(b * H_ + h) * U_) + u) * C_ + c] = (bf16)q[i];
}

// softmax over M*K with on-the-fly log prior; entropy & max. One block / row.
__global__ __launch_bounds__(256) void k_attn_softmax(
    const float* __restrict__ score,  // [B,H,U,MK]
    const float* __restrict__ assign, // [B,M,U]
    const float* __restrict__ tw,     // [B,M,H,K]
    bf16* __restrict__ attn,          // [B,H,U,MK]
    float* __restrict__ ent, float* __restrict__ mx)
{
    __shared__ float red[256];
    int row = blockIdx.x;             // (b*H+h)*U + u
    int u = row % U_;
    int h = (row / U_) % H_;
    int b = row / (U_ * H_);
    const float invs = 0.0625f;       // 1/sqrt(C), C=256
    long off = (long)row * MK_;
    float v[8];
    float lmax = -3.4e38f;
    #pragma unroll
    for (int j = 0; j < 8; ++j) {
        int i = threadIdx.x + j * 256;       // i = m*K + k
        int m = i >> 2, k = i & 3;
        float lp = logf(fmaxf(assign[((long)b * M_ + m) * U_ + u], 1e-8f)) +
                   logf(fmaxf(tw[(((long)b * M_ + m) * H_ + h) * K_ + k], 1e-8f));
        float s = score[off + i] * invs + lp;
        v[j] = s;
        lmax = fmaxf(lmax, s);
    }
    lmax = block_max(lmax, red);
    float ls = 0.f;
    #pragma unroll
    for (int j = 0; j < 8; ++j) { v[j] = expf(v[j] - lmax); ls += v[j]; }
    float inv = 1.0f / block_sum(ls, red);
    float e = 0.f, pm = 0.f;
    #pragma unroll
    for (int j = 0; j < 8; ++j) {
        int i = threadIdx.x + j * 256;
        float p = v[j] * inv;
        attn[off + i] = (bf16)p;
        float f = fmaxf(p, 1e-8f);
        e -= f * logf(f);
        pm = fmaxf(pm, f);
    }
    e = block_sum(e, red);
    pm = block_max(pm, red);
    if (threadIdx.x == 0) { ent[row] = e; mx[row] = pm; }
}

// fuse_in f32 [ROWS_, FIN_]
__global__ __launch_bounds__(256) void k_concat_f(
    const float* __restrict__ query,   const float* __restrict__ context, // [B,H,U,C]
    const float* __restrict__ trace_h, const float* __restrict__ anchor_h,
    const float* __restrict__ old_h,   const float* __restrict__ ent,
    const float* __restrict__ mx,      const float* __restrict__ usage,
    const float* __restrict__ massc,   float* __restrict__ fin)
{
    int row = blockIdx.x;
    int h = row % H_;
    int u = (row / H_) % U_;
    int b = row / (H_ * U_);
    int bu = b * U_ + u;
    int bhu = (b * H_ + h) * U_ + u;
    long o = (long)row * FIN_;
    for (int i = threadIdx.x; i < FIN_; i += 256) {
        float v;
        if (i < 256)        v = query[(long)row * C_ + i];
        else if (i < 512)   v = context[(long)bhu * C_ + (i - 256)];
        else if (i < 768)   v = trace_h[(long)row * C_ + (i - 512)];
        else if (i < 1024)  v = anchor_h[(long)bu * C_ + (i - 768)];
        else if (i < 1280)  v = old_h[(long)row * C_ + (i - 1024)];
        else if (i == 1280) v = ent[bhu];
        else if (i == 1281) v = mx[bhu];
        else if (i == 1282) v = fminf(fmaxf(usage[bu * H_ + h], 0.f), 1.f);
        else                v = massc[bu * H_ + h];
        fin[o + i] = v;
    }
}

// prototype softmax: logits[R,P]/TEMP -> bf16 weights
__global__ __launch_bounds__(256) void k_proto_softmax(
    const float* __restrict__ logits, bf16* __restrict__ w)
{
    __shared__ float red[256];
    int row = blockIdx.x;
    long o = (long)row * P_;
    float x = logits[o + threadIdx.x] * (1.0f / 0.7f);   // /TEMP
    float m = block_max(x, red);
    float e = expf(x - m);
    float s = block_sum(e, red);
    w[o + threadIdx.x] = (bf16)(e / s);
}

// rmag = MAX_RES * tanh(hidden . Wm + bm)
__global__ __launch_bounds__(256) void k_rmag(
    const float* __restrict__ hidden, const float* __restrict__ Wm,
    const float* __restrict__ bm, float* __restrict__ rmag)
{
    __shared__ float red[256];
    int row = blockIdx.x;
    long o = (long)row * (2 * C_);
    float s = 0.f;
    #pragma unroll
    for (int j = 0; j < 2; ++j) {
        int i = threadIdx.x + j * 256;
        s += hidden[o + i] * Wm[i];
    }
    float d = block_sum(s, red);
    if (threadIdx.x == 0) rmag[row] = 0.35f * tanhf(d + bm[0]);
}

// out = delta + normalize(rdir) * rmag
__global__ __launch_bounds__(256) void k_final(
    const float* __restrict__ delta, const float* __restrict__ rdir,
    const float* __restrict__ rmag, float* __restrict__ out)
{
    __shared__ float red[256];
    int row = blockIdx.x;
    long o = (long)row * D_;
    float s2 = 0.f;
    #pragma unroll
    for (int j = 0; j < 3; ++j) {
        float v = rdir[o + threadIdx.x + j * 256];
        s2 += v * v;
    }
    float nrm = sqrtf(block_sum(s2, red));
    float scale = rmag[row] / fmaxf(nrm, 1e-12f);
    #pragma unroll
    for (int j = 0; j < 3; ++j) {
        int i = threadIdx.x + j * 256;
        out[o + i] = delta[o + i] + rdir[o + i] * scale;
    }
}

// ---------------------------------------------------------------------------
// Host orchestration
// ---------------------------------------------------------------------------
static inline int gblk(long n) {
    long b = (n + 255) / 256;
    return (int)(b > 262144 ? 262144 : b);
}

extern "C" void kernel_launch(void* const* d_in, const int* in_sizes, int n_in,
                              void* d_out, int out_size, void* d_ws, size_t ws_size,
                              hipStream_t stream) {
    (void)in_sizes; (void)n_in; (void)out_size; (void)ws_size;

    // ---- inputs (setup_inputs dict order; params flattened in insertion
    //      order with tuples expanded) ----
    const float* assign   = (const float*)d_in[0];   // [B,M,U]
    const float* ftrace   = (const float*)d_in[1];   // [B,M,H,T]
    const float* topkraw  = (const float*)d_in[2];   // [B,M,H,K,D]
    const float* topkw    = (const float*)d_in[3];   // [B,M,H,K]
    const float* obssem   = (const float*)d_in[4];   // [B,M,4,D]
    const float* unitmem  = (const float*)d_in[5];   // [B,U,H,D]
    const float* unitconf = (const float*)d_in[6];   // [B,U]
    const float* unituse  = (const float*)d_in[7];   // [B,U,H]
    const float* psu      = (const float*)d_in[8];   // [B,M,H]
    int i9 = 9;
    const float* tr_g = (const float*)d_in[i9+0];
    const float* tr_b = (const float*)d_in[i9+1];
    const float* tr_W = (const float*)d_in[i9+2];
    const float* tr_bb= (const float*)d_in[i9+3];
    const float* ky_g = (const float*)d_in[i9+4];
    const float* ky_b = (const float*)d_in[i9+5];
    const float* ky_W = (const float*)d_in[i9+6];
    const float* ky_bb= (const float*)d_in[i9+7];
    const float* vl_g = (const float*)d_in[i9+8];
    const float* vl_b = (const float*)d_in[i9+9];
    const float* vl_W = (const float*)d_in[i9+10];
    const float* vl_bb= (const float*)d_in[i9+11];
    const float* sm_g = (const float*)d_in[i9+12];
    const float* sm_b = (const float*)d_in[i9+13];
    const float* sm_W = (const float*)d_in[i9+14];
    const float* sm_bb= (const float*)d_in[i9+15];
    const float* ol_g = (const float*)d_in[i9+16];
    const float* ol_b = (const float*)d_in[i9+17];
    const float* ol_W = (const float*)d_in[i9+18];
    const float* ol_bb= (const float*)d_in[i9+19];
    const float* qln_g= (const float*)d_in[i9+20];
    const float* qln_b= (const float*)d_in[i9+21];
    const float* qw1_W= (const float*)d_in[i9+22];
    const float* qw1_b= (const float*)d_in[i9+23];
    const float* qw2_W= (const float*)d_in[i9+24];
    const float* qw2_b= (const float*)d_in[i9+25];
    const float* fln_g= (const float*)d_in[i9+26];
    const float* fln_b= (const float*)d_in[i9+27];
    const float* fw1_W= (const float*)d_in[i9+28];
    const float* fw1_b= (const float*)d_in[i9+29];
    const float* fw2_W= (const float*)d_in[i9+30];
    const float* fw2_b= (const float*)d_in[i9+31];
    const float* pr_W = (const float*)d_in[i9+32];
    const float* pr_b = (const float*)d_in[i9+33];
    const float* rd_W = (const float*)d_in[i9+34];
    const float* rd_b = (const float*)d_in[i9+35];
    const float* rm_W = (const float*)d_in[i9+36];
    const float* rm_b = (const float*)d_in[i9+37];
    const float* codebook = (const float*)d_in[i9+38];
    float* out = (float*)d_out;

    // ---- workspace bump allocator (256B aligned) ----
    char* wsb = (char*)d_ws;
    size_t off = 0;
    auto alloc = [&](size_t bytes) -> void* {
        void* p = wsb + off;
        off += (bytes + 255) & ~(size_t)255;
        return p;
    };
    auto ab = [&](long n) { return (bf16*)alloc((size_t)n * 2); };
    auto af = [&](long n) { return (float*)alloc((size_t)n * 4); };

    // bf16 data buffers
    bf16* ftT_bf  = ab((long)B_*(H_*T_)*M_);    // [B, H*T, M]
    bf16* asT_bf  = ab((long)B_*U_*M_);         // [B, U, M]
    bf16* anT_bf  = ab((long)B_*D_*M_);         // [B, D, M]
    bf16* peT_bf  = ab((long)B_*(H_*D_)*M_);    // [B, H*D, M]
    bf16* Xkv     = ab((long)KVROWS_*D_);       // reused for key & value LN
    bf16* key_bf  = ab((long)BH_*MK_*C_);       // [B,H, MK, C]  (score B^T)
    bf16* valT_bf = ab((long)BH_*C_*MK_);       // [B,H, C, MK]  (context B^T)
    bf16* attn_bf = ab((long)BH_*U_*MK_);
    bf16* Xtr     = ab((long)ROWS_*T_);
    bf16* Xan     = ab((long)B_*U_*D_);
    bf16* Xev     = ab((long)ROWS_*D_);
    bf16* Xmem    = ab((long)ROWS_*D_);
    bf16* Xq      = ab((long)ROWS_*QINP_);
    bf16* Xf      = ab((long)ROWS_*FINP_);
    bf16* qh1     = ab((long)ROWS_*C_);
    bf16* qperm   = ab((long)BH_*U_*C_);
    bf16* fh1     = ab((long)ROWS_*2*C_);
    bf16* hid_bf  = ab((long)ROWS_*2*C_);
    bf16* wts_bf  = ab((long)ROWS_*P_);
    // bf16 weights (all stored transposed: [dout, din_padded])
    bf16* trWt  = ab((long)C_*T_);
    bf16* kyWt  = ab((long)C_*D_);
    bf16* vlWt  = ab((long)C_*D_);
    bf16* smWt  = ab((long)C_*D_);
    bf16* olWt  = ab((long)C_*D_);
    bf16* q1Wt  = ab((long)C_*QINP_);
    bf16* q2Wt  = ab((long)C_*C_);
    bf16* f1Wt  = ab((long)2*C_*FINP_);
    bf16* f2Wt  = ab((long)2*C_*2*C_);
    bf16* prWt  = ab((long)P_*2*C_);
    bf16* rdWt  = ab((long)D_*2*C_);
    bf16* cbWt  = ab((long)D_*P_);
    // f32 buffers
    float* unit_trace = af((long)ROWS_*T_);
    float* unit_evid  = af((long)ROWS_*D_);
    float* unit_anch  = af((long)B_*U_*D_);
    float* den_inv    = af(B_*U_);
    float* semu       = af((long)B_*U_*H_);
    float* massc      = af((long)B_*U_*H_);
    float* trace_h    = af((long)ROWS_*C_);
    float* anchor_h   = af((long)B_*U_*C_);
    float* evid_h     = af((long)ROWS_*C_);
    float* old_h      = af((long)ROWS_*C_);
    float* qin        = af((long)ROWS_*QIN_);
    float* query_f    = af((long)ROWS_*C_);
    float* score      = af((long)BH_*U_*MK_);
    float* entb       = af(ROWS_);
    float* mxb        = af(ROWS_);
    float* context    = af((long)BH_*U_*C_);
    float* fin        = af((long)ROWS_*FIN_);
    float* hidden_f   = af((long)ROWS_*2*C_);
    float* logits     = af((long)ROWS_*P_);
    float* delta      = af((long)ROWS_*D_);
    float* rdir       = af((long)ROWS_*D_);
    float* rmag       = af(ROWS_);

    dim3 blk(256);
    auto gemm = [&](const bf16* A, long sA, const bf16* Bt, long sB,
                    const float* bias, const float* rs, long sR,
                    float* Cf, long sCf, bf16* Cb, long sCb,
                    int Mr, int Nr, int Kr, int tCb, int gel, int batch) {
        dim3 g(Nr / 64, Mr / 64, batch);
        wmma_gemm<<<g, blk, 0, stream>>>(A, sA, Bt, sB, bias, rs, sR,
                                         Cf, sCf, Cb, sCb, Mr, Nr, Kr, tCb, gel);
    };

    // ---- stage 0: conversions & prep ----
    k_ftT<<<gblk((long)B_*M_*H_*T_), blk, 0, stream>>>(ftrace, ftT_bf);
    k_assignT<<<gblk((long)B_*U_*M_), blk, 0, stream>>>(assign, asT_bf);
    k_anchorT<<<gblk((long)B_*M_*D_), blk, 0, stream>>>(obssem, anT_bf);
    k_point_evidenceT<<<gblk((long)B_*M_*H_*D_), blk, 0, stream>>>(topkraw, topkw, peT_bf);
    k_pools<<<dim3(B_*U_), blk, 0, stream>>>(assign, psu, den_inv, semu, massc);
    // weights -> transposed bf16
    k_pack_wT<<<gblk((long)C_*T_), blk, 0, stream>>>(tr_W, trWt, T_, C_, T_);
    k_pack_wT<<<gblk((long)C_*D_), blk, 0, stream>>>(ky_W, kyWt, D_, C_, D_);
    k_pack_wT<<<gblk((long)C_*D_), blk, 0, stream>>>(vl_W, vlWt, D_, C_, D_);
    k_pack_wT<<<gblk((long)C_*D_), blk, 0, stream>>>(sm_W, smWt, D_, C_, D_);
    k_pack_wT<<<gblk((long)C_*D_), blk, 0, stream>>>(ol_W, olWt, D_, C_, D_);
    k_pack_wT<<<gblk((long)C_*QINP_), blk, 0, stream>>>(qw1_W, q1Wt, QIN_, C_, QINP_);
    k_pack_wT<<<gblk((long)C_*C_), blk, 0, stream>>>(qw2_W, q2Wt, C_, C_, C_);
    k_pack_wT<<<gblk((long)2*C_*FINP_), blk, 0, stream>>>(fw1_W, f1Wt, FIN_, 2*C_, FINP_);
    k_pack_wT<<<gblk((long)2*C_*2*C_), blk, 0, stream>>>(fw2_W, f2Wt, 2*C_, 2*C_, 2*C_);
    k_pack_wT<<<gblk((long)P_*2*C_), blk, 0, stream>>>(pr_W, prWt, 2*C_, P_, 2*C_);
    k_pack_wT<<<gblk((long)D_*2*C_), blk, 0, stream>>>(rd_W, rdWt, 2*C_, D_, 2*C_);
    k_pack_wT<<<gblk((long)D_*P_), blk, 0, stream>>>(codebook, cbWt, P_, D_, P_);

    // ---- stage 1: pooling GEMMs (A = assignT [U,M], row-scale 1/den) ----
    gemm(asT_bf, (long)U_*M_, ftT_bf, (long)(H_*T_)*M_, nullptr, den_inv, U_,
         unit_trace, (long)U_*H_*T_, nullptr, 0, U_, H_*T_, M_, 0, 0, B_);
    gemm(asT_bf, (long)U_*M_, peT_bf, (long)(H_*D_)*M_, nullptr, den_inv, U_,
         unit_evid, (long)U_*H_*D_, nullptr, 0, U_, H_*D_, M_, 0, 0, B_);
    gemm(asT_bf, (long)U_*M_, anT_bf, (long)D_*M_, nullptr, den_inv, U_,
         unit_anch, (long)U_*D_, nullptr, 0, U_, D_, M_, 0, 0, B_);

    // ---- stage 2: LN + projections ----
    k_ln_rows<<<dim3(ROWS_), blk, 0, stream>>>(unit_trace, tr_g, tr_b, Xtr, T_, T_, 0);
    gemm(Xtr, 0, trWt, 0, tr_bb, nullptr, 0, trace_h, 0, nullptr, 0,
         ROWS_, C_, T_, 0, 0, 1);
    k_ln_rows<<<dim3(B_*U_), blk, 0, stream>>>(unit_anch, sm_g, sm_b, Xan, D_, D_, 0);
    gemm(Xan, 0, smWt, 0, sm_bb, nullptr, 0, anchor_h, 0, nullptr, 0,
         B_*U_, C_, D_, 0, 0, 1);
    k_ln_rows<<<dim3(ROWS_), blk, 0, stream>>>(unit_evid, sm_g, sm_b, Xev, D_, D_, 0);
    gemm(Xev, 0, smWt, 0, sm_bb, nullptr, 0, evid_h, 0, nullptr, 0,
         ROWS_, C_, D_, 0, 0, 1);
    k_ln_rows<<<dim3(ROWS_), blk, 0, stream>>>(unitmem, ol_g, ol_b, Xmem, D_, D_, 0);
    gemm(Xmem, 0, olWt, 0, ol_bb, nullptr, 0, old_h, 0, nullptr, 0,
         ROWS_, C_, D_, 0, 0, 1);

    // ---- stage 3: query MLP ----
    k_concat_q<<<dim3(ROWS_), blk, 0, stream>>>(trace_h, anchor_h, evid_h, old_h,
                                                unitconf, unituse, semu, qin);
    k_ln_rows<<<dim3(ROWS_), blk, 0, stream>>>(qin, qln_g, qln_b, Xq, QIN_, QINP_, 0);
    gemm(Xq, 0, q1Wt, 0, qw1_b, nullptr, 0, nullptr, 0, qh1, 0,
         ROWS_, C_, QINP_, 0, 1, 1);                       // GELU -> bf16
    gemm(qh1, 0, q2Wt, 0, qw2_b, nullptr, 0, query_f, 0, nullptr, 0,
         ROWS_, C_, C_, 0, 0, 1);
    k_permute_query<<<gblk((long)ROWS_*C_), blk, 0, stream>>>(query_f, qperm);

    // ---- stage 4: key/value projections ([B,H,MK,*] layout) ----
    k_ln_rows<<<dim3(KVROWS_), blk, 0, stream>>>(topkraw, ky_g, ky_b, Xkv, D_, D_, 1);
    gemm(Xkv, (long)MK_*D_, kyWt, 0, ky_bb, nullptr, 0, nullptr, 0,
         key_bf, (long)MK_*C_, MK_, C_, D_, 0, 0, BH_);
    k_ln_rows<<<dim3(KVROWS_), blk, 0, stream>>>(topkraw, vl_g, vl_b, Xkv, D_, D_, 1);
    gemm(Xkv, (long)MK_*D_, vlWt, 0, vl_bb, nullptr, 0, nullptr, 0,
         valT_bf, (long)C_*MK_, MK_, C_, D_, 1, 0, BH_);   // bf16 out transposed

    // ---- stage 5: attention ----
    gemm(qperm, (long)U_*C_, key_bf, (long)MK_*C_, nullptr, nullptr, 0,
         score, (long)U_*MK_, nullptr, 0, U_, MK_, C_, 0, 0, BH_);
    k_attn_softmax<<<dim3(ROWS_), blk, 0, stream>>>(score, assign, topkw,
                                                    attn_bf, entb, mxb);
    gemm(attn_bf, (long)U_*MK_, valT_bf, (long)C_*MK_, nullptr, nullptr, 0,
         context, (long)U_*C_, nullptr, 0, U_, C_, MK_, 0, 0, BH_);

    // ---- stage 6: fuse MLP ----
    k_concat_f<<<dim3(ROWS_), blk, 0, stream>>>(query_f, context, trace_h, anchor_h,
                                                old_h, entb, mxb, unituse, massc, fin);
    k_ln_rows<<<dim3(ROWS_), blk, 0, stream>>>(fin, fln_g, fln_b, Xf, FIN_, FINP_, 0);
    gemm(Xf, 0, f1Wt, 0, fw1_b, nullptr, 0, nullptr, 0, fh1, 0,
         ROWS_, 2*C_, FINP_, 0, 1, 1);                     // GELU -> bf16
    gemm(fh1, 0, f2Wt, 0, fw2_b, nullptr, 0, hidden_f, 0, hid_bf, 0,
         ROWS_, 2*C_, 2*C_, 0, 1, 1);                      // GELU -> f32 + bf16

    // ---- stage 7: heads ----
    gemm(hid_bf, 0, prWt, 0, pr_b, nullptr, 0, logits, 0, nullptr, 0,
         ROWS_, P_, 2*C_, 0, 0, 1);
    k_proto_softmax<<<dim3(ROWS_), blk, 0, stream>>>(logits, wts_bf);
    gemm(wts_bf, 0, cbWt, 0, nullptr, nullptr, 0, delta, 0, nullptr, 0,
         ROWS_, D_, P_, 0, 0, 1);
    gemm(hid_bf, 0, rdWt, 0, rd_b, nullptr, 0, rdir, 0, nullptr, 0,
         ROWS_, D_, 2*C_, 0, 0, 1);
    k_rmag<<<dim3(ROWS_), blk, 0, stream>>>(hidden_f, rm_W, rm_b, rmag);
    k_final<<<dim3(ROWS_), blk, 0, stream>>>(delta, rdir, rmag, out);
}
